// PhotonicBasisActivationLayerIntervalAffineClean_79748952752386
// MI455X (gfx1250) — matrix operand order repaired and checked by reading, not compiled, not run on hardware
//
#include <hip/hip_runtime.h>

// ---------------------------------------------------------------------------
// Photonic basis activation + einsum('bik,oik->bo') fused GEMM for gfx1250.
//
//   u[b,i] = alpha[i]*x[b,i] + beta[i]
//   f_k(u) = b1*log(1 + b2*log1p(expm1(b3*u)^b4)) + b5*u + b6*u^2 + b7*u^3 + b8*u^4
//   y[b,o] = sum_{i,k} f_k(u[b,i]) * coeffs[o,i,k]
//
// GEMM view: A = basis (32768 x 4096, computed on the fly, bf16),
//            B = coeffs reshaped (4096 x 32, bf16 from L2),
//            C = y (32768 x 32, f32) via v_wmma_f32_16x16x32_bf16.
//
// Bottleneck is the 134M basis evals (5 transcendentals each). We force the
// native v_exp_f32 / v_log_f32 (base-2) TRANS instructions via
// __builtin_amdgcn_exp2f / __builtin_amdgcn_logf and pre-fold all ln2/log2e
// conversion factors into the per-k coefficient table in LDS, so each eval is
// exactly 2x v_exp_f32 + 3x v_log_f32 + ~9 FMAs. WMMA (also TRANS-class)
// interleaves with this stream; DRAM traffic (~38 MB) is negligible at
// 23.3 TB/s.
// ---------------------------------------------------------------------------

typedef __attribute__((ext_vector_type(16))) __bf16 v16bf;
typedef __attribute__((ext_vector_type(8)))  float  v8f;
typedef __attribute__((ext_vector_type(4)))  float  v4f;
typedef __attribute__((ext_vector_type(2)))  float  v2f;

#define B_ROWS 32768
#define I_DIM  256
#define O_DIM  32
#define K_DIM  16
#define KTOT   4096   // I_DIM * K_DIM (GEMM reduction length)
#define NCHUNK 128    // KTOT / 32 (bf16 WMMA K=32 per step)

#define LN2f    0.6931471805599453f
#define LOG2Ef  1.4426950408889634f

// One basis evaluation on pre-folded coefficients:
//   c0 = b1*ln2, c1 = b2*ln2, c2 = b3*log2e, c3 = b4, c4..c7 = b5..b8
// exp(x) = exp2(x*log2e), log(x) = log2(x)*ln2 — all folds absorbed into c*.
// Emits exactly: 2x v_exp_f32, 3x v_log_f32, plus FMAs. All args in safe
// range (b3*u in [2.5e-4, 1.2] > 0, so t > 0, pw > 0).
__device__ __forceinline__ float photonic_eval(float u, const float* __restrict__ c) {
    float t    = __builtin_amdgcn_exp2f(c[2] * u) - 1.0f;            // expm1(b3*u)
    float pw   = __builtin_amdgcn_exp2f(c[3] * __builtin_amdgcn_logf(t)); // t^b4
    float il   = __builtin_amdgcn_logf(1.0f + pw);                   // log1p scaled
    float lo   = __builtin_amdgcn_logf(fmaf(c[1], il, 1.0f));        // outer log
    float poly = u * fmaf(u, fmaf(u, fmaf(u, c[7], c[6]), c[5]), c[4]);
    return fmaf(c[0], lo, poly);
}

__global__ __launch_bounds__(256)
void photonic_wmma_kernel(const float* __restrict__ x,
                          const float* __restrict__ coeffs,
                          const float* __restrict__ b_coef,
                          const float* __restrict__ alpha,
                          const float* __restrict__ beta,
                          float* __restrict__ out) {
    __shared__ float s_bcoef[K_DIM][8];   // pre-folded b1..b8 per k
    __shared__ float s_alpha[I_DIM];
    __shared__ float s_beta[I_DIM];

    const int tid = threadIdx.x;
    if (tid < K_DIM * 8) {
        const int j = tid & 7;
        float v = b_coef[tid];
        if (j == 0) v *= LN2f;      // b1 -> b1*ln2  (log(outer) = ln2*log2)
        if (j == 1) v *= LN2f;      // b2 -> b2*ln2  (log1p    = ln2*log2)
        if (j == 2) v *= LOG2Ef;    // b3 -> b3*log2e (expm1 via exp2)
        s_bcoef[tid >> 3][j] = v;
    }
    s_alpha[tid] = alpha[tid];
    s_beta[tid]  = beta[tid];
    __syncthreads();

    const int lane  = tid & 31;
    const int wave  = tid >> 5;               // 0..7  -> 16-row strip within block
    const int mlane = lane & 15;              // A row / B column / C column
    const int khalf = lane >> 4;              // 0: k 0..7 | K 0..7,16..23 ; 1: k 8..15 | K 8..15,24..31
    const int kofs  = khalf * 8;              // basis-k offset this lane covers in A
    const int bko   = khalf * 8;              // K-row offset this lane covers in B (K 8h..8h+7, 16+8h..16+8h+7)

    const int row0 = blockIdx.x * 128 + wave * 16;
    const int m    = row0 + mlane;
    const float* xrow = x + (size_t)m * I_DIM;

    // B-matrix columns this lane feeds (N-tile 0: o = mlane, N-tile 1: o = 16+mlane).
    const float* bc0 = coeffs + (size_t)(mlane)      * KTOT + bko;
    const float* bc1 = coeffs + (size_t)(16 + mlane) * KTOT + bko;

    v8f acc0 = {};
    v8f acc1 = {};

    for (int c = 0; c < NCHUNK; ++c) {
        const int i0 = c * 2;

        // ---- A fragment: 16 basis evals per lane (8 k's x 2 i's), bf16 ----
        v2f xv = *(const v2f*)(xrow + i0);
        float u0 = fmaf(s_alpha[i0],     xv.x, s_beta[i0]);
        float u1 = fmaf(s_alpha[i0 + 1], xv.y, s_beta[i0 + 1]);

        v16bf a;
#pragma unroll
        for (int e = 0; e < 8; ++e) {
            const float* cf = s_bcoef[kofs + e];   // broadcast LDS read per half-wave
            a[e]     = (__bf16)photonic_eval(u0, cf);
            a[e + 8] = (__bf16)photonic_eval(u1, cf);
        }

        // ---- B fragments: f32 coeffs (L2-resident) -> bf16 on the fly ----
        const float* p0 = bc0 + c * 32;
        const float* p1 = bc1 + c * 32;
        v16bf b0, b1;
#pragma unroll
        for (int e = 0; e < 16; e += 4) {
            v4f f0 = *(const v4f*)(p0 + e + (e & 8));
            v4f f1 = *(const v4f*)(p1 + e + (e & 8));
            b0[e] = (__bf16)f0.x; b0[e+1] = (__bf16)f0.y;
            b0[e+2] = (__bf16)f0.z; b0[e+3] = (__bf16)f0.w;
            b1[e] = (__bf16)f1.x; b1[e+1] = (__bf16)f1.y;
            b1[e+2] = (__bf16)f1.z; b1[e+3] = (__bf16)f1.w;
        }

        // D = A x B + C  (16x16x32 bf16 -> f32), A shared across both N-tiles.
        acc0 = __builtin_amdgcn_wmma_f32_16x16x32_bf16(
                   false, a, false, b0, (short)0, acc0, false, false);
        acc1 = __builtin_amdgcn_wmma_f32_16x16x32_bf16(
                   false, a, false, b1, (short)0, acc1, false, false);
    }

    // ---- Store C/D per ISA layout: VGPR r, lanes 0-15 -> M=r, lanes 16-31 -> M=r+8 ----
    const int n   = mlane;
    const int mhi = khalf * 8;
#pragma unroll
    for (int r = 0; r < 8; ++r) {
        const int mm = row0 + mhi + r;
        out[(size_t)mm * O_DIM + n]      = acc0[r];
        out[(size_t)mm * O_DIM + 16 + n] = acc1[r];
    }
}

extern "C" void kernel_launch(void* const* d_in, const int* in_sizes, int n_in,
                              void* d_out, int out_size, void* d_ws, size_t ws_size,
                              hipStream_t stream) {
    const float* x      = (const float*)d_in[0];   // (B, I)
    const float* coeffs = (const float*)d_in[1];   // (O, I, K) == (O, 4096)
    const float* b_coef = (const float*)d_in[2];   // (K, 8)
    const float* alpha  = (const float*)d_in[3];   // (I,)
    const float* beta   = (const float*)d_in[4];   // (I,)
    float* out = (float*)d_out;                    // (B, O) f32

    (void)in_sizes; (void)n_in; (void)out_size; (void)d_ws; (void)ws_size;

    dim3 grid(B_ROWS / 128);
    dim3 block(256);
    hipLaunchKernelGGL(photonic_wmma_kernel, grid, block, 0, stream,
                       x, coeffs, b_coef, alpha, beta, out);
}
